// Model_58643483460201
// MI455X (gfx1250) — compile-verified, matrix operand order checked by reading, not run on hardware
//
#include <hip/hip_runtime.h>

// ---------------------------------------------------------------------------
// Problem constants (from the reference)
// ---------------------------------------------------------------------------
constexpr int  N_    = 2048;
constexpr int  SHOT_ = 5;
constexpr int  H_    = 8;
constexpr int  D_    = 256;
constexpr int  L_    = 3;
constexpr int  FFN_  = 8192;
constexpr int  TD_   = H_ * D_;          // 2048
constexpr int  S_    = 2 * SHOT_ + 1;    // 11
constexpr int  P_    = SHOT_ + 2;        // 7
constexpr long M_    = (long)N_ * S_;    // 22528 rows (divisible by 128)
constexpr float EPS_   = 1e-5f;
constexpr float SCALE_ = 0.0441941738241592f;  // (D*2)^-0.5 = 512^-0.5

typedef __bf16 bf16;
typedef __attribute__((ext_vector_type(16))) __bf16 v16bf;
typedef __attribute__((ext_vector_type(8)))  float  v8f;
typedef int v4i __attribute__((vector_size(16)));   // matches builtin param spelling

#if defined(__gfx1250__) && __has_builtin(__builtin_amdgcn_global_load_async_to_lds_b128)
#define ASYNC_LDS 1
typedef __attribute__((address_space(1))) v4i* as1_v4i_p;
typedef __attribute__((address_space(3))) v4i* as3_v4i_p;
#else
#define ASYNC_LDS 0
#endif

__device__ inline bf16 f2bf(float f) {
  unsigned u = __float_as_uint(f);
  u = (u + 0x7FFFu + ((u >> 16) & 1u)) >> 16;
  unsigned short h = (unsigned short)u;
  bf16 r;
  __builtin_memcpy(&r, &h, 2);
  return r;
}

__device__ inline float gelu_exact(float v) {
  return 0.5f * v * (1.0f + erff(v * 0.7071067811865475f));
}

// 16B memory -> LDS copy: async (ASYNCcnt-tracked, no VGPR round trip) if available.
__device__ inline void copy16_to_lds(bf16* __restrict__ ls, const bf16* __restrict__ gs) {
#if ASYNC_LDS
  __builtin_amdgcn_global_load_async_to_lds_b128(
      (as1_v4i_p)(unsigned long long)(const void*)gs,
      (as3_v4i_p)(unsigned)(unsigned long long)(void*)ls,
      0, 0);
#else
  *reinterpret_cast<uint4*>(ls) = *reinterpret_cast<const uint4*>(gs);
#endif
}

__device__ inline void wait_async_lds() {
#if ASYNC_LDS
#if __has_builtin(__builtin_amdgcn_s_wait_asynccnt)
  __builtin_amdgcn_s_wait_asynccnt(0);
#else
  asm volatile("s_wait_asynccnt 0" ::: "memory");
#endif
#endif
}

// ---------------------------------------------------------------------------
// fp32 -> bf16 conversion (grid-stride)
// ---------------------------------------------------------------------------
__global__ void k_f32_to_bf16(const float* __restrict__ in, bf16* __restrict__ out, long n) {
  long i = (long)blockIdx.x * blockDim.x + threadIdx.x;
  long stride = (long)gridDim.x * blockDim.x;
  for (; i < n; i += stride) out[i] = f2bf(in[i]);
}

// ---------------------------------------------------------------------------
// Build x0: x[n, s, h*D + d] = (s==0 ? cls[d] : entity[n, s-1, d]) (tiled over H)
// ---------------------------------------------------------------------------
__global__ void k_build_x(const float* __restrict__ ent, const float* __restrict__ cls,
                          float* __restrict__ x) {
  long idx = (long)blockIdx.x * 256 + threadIdx.x;   // grid sized exactly N*S*TD/256
  int  j   = (int)(idx % TD_);
  long ns  = idx / TD_;
  int  s   = (int)(ns % S_);
  long n   = ns / S_;
  int  d   = j & (D_ - 1);
  x[idx] = (s == 0) ? cls[d] : ent[(n * (S_ - 1) + (s - 1)) * D_ + d];
}

// ---------------------------------------------------------------------------
// LN of pair_pos_w rows (7 x 256): one block per row, 256 threads
// ---------------------------------------------------------------------------
__global__ __launch_bounds__(256) void k_wln(const float* __restrict__ w,
                                             const float* __restrict__ g,
                                             const float* __restrict__ b,
                                             float* __restrict__ out) {
  __shared__ float red[256];
  int p = blockIdx.x;
  float v = w[p * D_ + threadIdx.x];
  red[threadIdx.x] = v;
  __syncthreads();
  for (int off = 128; off > 0; off >>= 1) {
    if (threadIdx.x < off) red[threadIdx.x] += red[threadIdx.x + off];
    __syncthreads();
  }
  float mean = red[0] / D_;
  __syncthreads();
  float dd = v - mean;
  red[threadIdx.x] = dd * dd;
  __syncthreads();
  for (int off = 128; off > 0; off >>= 1) {
    if (threadIdx.x < off) red[threadIdx.x] += red[threadIdx.x + off];
    __syncthreads();
  }
  float inv = rsqrtf(red[0] / D_ + EPS_);
  out[p * D_ + threadIdx.x] = dd * inv * g[threadIdx.x] + b[threadIdx.x];
}

// ---------------------------------------------------------------------------
// pos projections: pq/pk = tile(wln,(1,H)) @ W.T + b  (tiny: 2*7*2048 dots)
// ---------------------------------------------------------------------------
__global__ void k_posproj(const float* __restrict__ wln,
                          const float* __restrict__ qw, const float* __restrict__ qb,
                          const float* __restrict__ kw, const float* __restrict__ kb,
                          float* __restrict__ pq, float* __restrict__ pk) {
  int o     = blockIdx.x * 256 + threadIdx.x;     // grid sized exactly 2*P*TD/256
  int which = o / (P_ * TD_);
  int rem   = o - which * (P_ * TD_);
  int p     = rem / TD_;
  int j     = rem - p * TD_;
  const float* W    = which ? kw : qw;
  const float* rowl = wln + p * D_;
  float acc = 0.f;
  for (int t = 0; t < TD_; ++t) acc += rowl[t & (D_ - 1)] * W[(long)j * TD_ + t];
  acc += which ? kb[j] : qb[j];
  if (!which) acc *= SCALE_;
  (which ? pk : pq)[rem] = acc;
}

// ---------------------------------------------------------------------------
// Positional attention bias (H x 11 x 11)
// ---------------------------------------------------------------------------
__global__ __launch_bounds__(128) void k_bias(const float* __restrict__ pq,
                                              const float* __restrict__ pk,
                                              float* __restrict__ bias) {
  __shared__ float apb[P_][P_];
  int h = blockIdx.x;
  if (threadIdx.x < P_ * P_) {
    int p = threadIdx.x / P_, q = threadIdx.x % P_;
    float acc = 0.f;
    for (int d = 0; d < D_; ++d)
      acc += pq[(long)p * TD_ + h * D_ + d] * pk[(long)q * TD_ + h * D_ + d];
    apb[p][q] = acc;
  }
  __syncthreads();
  if (threadIdx.x < S_ * S_) {
    int s = threadIdx.x / S_, t = threadIdx.x % S_;
    float val;
    if (s == 0)       val = apb[0][0];                       // cls2other
    else if (t == 0)  val = apb[1][1];                       // other2cls
    else              val = apb[1 + (s + 1) / 2][1 + (t + 1) / 2];
    bias[(h * S_ + s) * S_ + t] = val;
  }
}

// ---------------------------------------------------------------------------
// bf16 NT GEMM via v_wmma_f32_16x16x32_bf16:
//   C[M,Nc] = A[M,K] * B[Nc,K]^T + bias, optional GELU, optional bf16 out.
// Block 128x128, K-step 32, 8 waves each owning a 64x32 tile (4x2 frags).
// Double-buffered LDS; next K-tile is copied with GLOBAL_LOAD_ASYNC_TO_LDS
// while the current tile is consumed by WMMA; s_wait_asynccnt before the
// tile-swap barrier. All dims divide tile sizes -> EXEC all-ones at WMMA.
// ---------------------------------------------------------------------------
constexpr int BM = 128, BN = 128, BK = 32, LDT = 40;  // LDT: padded LDS stride (bf16)

union FragU { unsigned u[8]; v16bf v; };

// 16-bit A/B 16x32 fragment layout (ISA 7.12.2): lane L holds row/col L%16,
// VGPR j: K = (j%4)*2 + (j/4)*16 + (L/16)*8
__device__ inline void load_frag_lds(const bf16* __restrict__ base, int row0, int lane,
                                     FragU& f) {
  int rc = row0 + (lane & 15);
  const bf16* p = base + rc * LDT + ((lane >> 4) << 3);
#pragma unroll
  for (int j = 0; j < 8; ++j) {
    int k = ((j & 3) << 1) + ((j >> 2) << 4);
    f.u[j] = *reinterpret_cast<const unsigned*>(p + k);
  }
}

template <int ACT, int OUTBF>
__global__ __launch_bounds__(256, 2) void k_gemm_bf16nt(
    const bf16* __restrict__ A, const bf16* __restrict__ B,
    const float* __restrict__ bias, void* __restrict__ Cout,
    int Nc, int K) {
  __shared__ bf16 As[2][BM * LDT];
  __shared__ bf16 Bs[2][BN * LDT];
  const int tid  = threadIdx.x;
  const int lane = tid & 31;
  const int wid  = tid >> 5;
  const int wm   = (wid & 1) << 6;   // 0 / 64
  const int wn   = (wid >> 1) << 5;  // 0 / 32 / 64 / 96
  const long m0  = (long)blockIdx.y * BM;
  const long n0  = (long)blockIdx.x * BN;

  v8f acc[4][2] = {};

  const int lr = tid >> 2;          // 0..63
  const int lc = (tid & 3) << 3;    // 0,8,16,24 (bf16 elems; 16B chunks)

  auto issue_tile = [&](int k0, int buf) {
#pragma unroll
    for (int it = 0; it < 2; ++it) {
      int r = lr + (it << 6);
      copy16_to_lds(&As[buf][r * LDT + lc], A + (m0 + r) * K + (k0 + lc));
      copy16_to_lds(&Bs[buf][r * LDT + lc], B + (n0 + r) * K + (k0 + lc));
    }
  };

  // Prologue: fill buffer 0.
  issue_tile(0, 0);
  wait_async_lds();
  __syncthreads();

  int buf = 0;
  for (int k0 = 0; k0 < K; k0 += BK) {
    // Kick off the next tile's async copies into the other buffer.
    if (k0 + BK < K) {
      issue_tile(k0 + BK, buf ^ 1);
      if (k0 + 2 * BK < K) {  // warm L2 two tiles ahead (global_prefetch_b8)
        __builtin_prefetch(A + (m0 + lr) * K + (k0 + 2 * BK + lc), 0, 1);
        __builtin_prefetch(B + (n0 + lr) * K + (k0 + 2 * BK + lc), 0, 1);
      }
    }

    // Consume current buffer with WMMA while the async copies fly.
    FragU af[4], bfg[2];
#pragma unroll
    for (int mi = 0; mi < 4; ++mi) load_frag_lds(As[buf], wm + (mi << 4), lane, af[mi]);
#pragma unroll
    for (int ni = 0; ni < 2; ++ni) load_frag_lds(Bs[buf], wn + (ni << 4), lane, bfg[ni]);
#pragma unroll
    for (int mi = 0; mi < 4; ++mi)
#pragma unroll
      for (int ni = 0; ni < 2; ++ni)
        acc[mi][ni] = __builtin_amdgcn_wmma_f32_16x16x32_bf16(
            false, af[mi].v, false, bfg[ni].v, (short)0, acc[mi][ni], false, false);

    wait_async_lds();   // this wave's copies into buf^1 have landed
    __syncthreads();    // every wave is done reading buf
    buf ^= 1;
  }

  // C layout (ISA 7.12.2): VGPR i -> M = i + (lane>>4)*8, N = lane&15
  const int cn = lane & 15;
  const int mb = (lane >> 4) << 3;
#pragma unroll
  for (int mi = 0; mi < 4; ++mi) {
#pragma unroll
    for (int ni = 0; ni < 2; ++ni) {
      long col = n0 + wn + (ni << 4) + cn;
      float bb = bias ? bias[col] : 0.f;
#pragma unroll
      for (int i = 0; i < 8; ++i) {
        long row = m0 + wm + (mi << 4) + mb + i;
        float v  = acc[mi][ni][i] + bb;
        if (ACT == 1) v = gelu_exact(v);
        if (OUTBF) ((bf16*)Cout)[row * Nc + col] = f2bf(v);
        else       ((float*)Cout)[row * Nc + col] = v;
      }
    }
  }
}

// ---------------------------------------------------------------------------
// Attention for one (n, h): RoPE on q/k/v, 11x11 dual softmax, 11x256 output.
// ---------------------------------------------------------------------------
__global__ __launch_bounds__(128) void k_attention(
    const float* __restrict__ q, const float* __restrict__ k, const float* __restrict__ v,
    const float* __restrict__ bias, float* __restrict__ o) {
  __shared__ float qs[S_][D_], ks[S_][D_], vs[S_][D_];
  __shared__ float pr[S_][S_ + 1];
  __shared__ float sc[S_][S_ + 1];
  int n = blockIdx.x, h = blockIdx.y;

  for (int idx = threadIdx.x; idx < S_ * D_; idx += 128) {
    int s = idx >> 8, d = idx & (D_ - 1);
    long base = ((long)(n * S_ + s)) * TD_ + h * D_;
    int  i    = d >> 1;
    int  role = (s == 0) ? 0 : ((s & 1) ? 1 : 2);
    float ang = (float)role * __powf(10000.f, -(float)i * (1.f / 128.f));
    float ci = __cosf(ang), si = __sinf(ang);
    int dp = d ^ 1;
    float qv = q[base + d], qp = q[base + dp];
    float kv = k[base + d], kp = k[base + dp];
    float vv = v[base + d], vp = v[base + dp];
    float sgn = (d & 1) ? 1.f : -1.f;       // rotate_half: even -> -x[odd], odd -> x[even]
    qs[s][d] = qv * ci + sgn * qp * si;
    ks[s][d] = kv * ci + sgn * kp * si;
    vs[s][d] = vv * ci + sgn * vp * si;
  }
  __syncthreads();

  for (int p = threadIdx.x; p < S_ * S_; p += 128) {
    int s = p / S_, t = p - s * S_;
    float acc = 0.f;
    for (int d = 0; d < D_; ++d) acc += qs[s][d] * ks[t][d];
    sc[s][t] = acc * SCALE_ + bias[(h * S_ + s) * S_ + t];
  }
  __syncthreads();

  if (threadIdx.x < S_) {
    int s = threadIdx.x;
    float e1[S_], e2[S_];
    float mx1 = -1e30f, mx2 = -1e30f;
    for (int t = 0; t < S_; ++t) {
      float sv = sc[s][t];
      bool allowed = (s == 0) || ((t > 0) && (((s - 1) >> 1) == ((t - 1) >> 1)));
      float mv = allowed ? sv : -1e9f;
      mx1 = fmaxf(mx1, sv);
      mx2 = fmaxf(mx2, mv);
      e1[t] = sv;  e2[t] = mv;
    }
    float s1 = 0.f, s2 = 0.f;
    for (int t = 0; t < S_; ++t) {
      e1[t] = __expf(e1[t] - mx1); s1 += e1[t];
      e2[t] = __expf(e2[t] - mx2); s2 += e2[t];
    }
    float r1 = 1.f / s1, r2 = 1.f / s2;
    for (int t = 0; t < S_; ++t) pr[s][t] = 0.5f * (e1[t] * r1 + e2[t] * r2);
  }
  __syncthreads();

  for (int idx = threadIdx.x; idx < S_ * D_; idx += 128) {
    int s = idx >> 8, d = idx & (D_ - 1);
    float acc = 0.f;
    for (int t = 0; t < S_; ++t) acc += pr[s][t] * vs[t][d];
    o[((long)(n * S_ + s)) * TD_ + h * D_ + d] = acc;
  }
}

// ---------------------------------------------------------------------------
// x = LayerNorm(res + c) * g + b   (one block per row of 2048)
// ---------------------------------------------------------------------------
__global__ __launch_bounds__(256) void k_add_ln(const float* __restrict__ res,
                                               const float* __restrict__ c,
                                               const float* __restrict__ g,
                                               const float* __restrict__ b,
                                               float* __restrict__ out) {
  __shared__ float red[256];
  long row = blockIdx.x;
  const float* pr = res + row * TD_;
  const float* pc = c + row * TD_;
  float vals[8];
  float s = 0.f;
#pragma unroll
  for (int i = 0; i < 8; ++i) {
    int j = threadIdx.x + i * 256;
    vals[i] = pr[j] + pc[j];
    s += vals[i];
  }
  red[threadIdx.x] = s;
  __syncthreads();
  for (int off = 128; off > 0; off >>= 1) {
    if (threadIdx.x < off) red[threadIdx.x] += red[threadIdx.x + off];
    __syncthreads();
  }
  float mean = red[0] / TD_;
  __syncthreads();
  float vsum = 0.f;
#pragma unroll
  for (int i = 0; i < 8; ++i) { float d = vals[i] - mean; vsum += d * d; }
  red[threadIdx.x] = vsum;
  __syncthreads();
  for (int off = 128; off > 0; off >>= 1) {
    if (threadIdx.x < off) red[threadIdx.x] += red[threadIdx.x + off];
    __syncthreads();
  }
  float inv = rsqrtf(red[0] / TD_ + EPS_);
#pragma unroll
  for (int i = 0; i < 8; ++i) {
    int j = threadIdx.x + i * 256;
    out[row * TD_ + j] = (vals[i] - mean) * inv * g[j] + b[j];
  }
}

// ---------------------------------------------------------------------------
// out[n, :] = x[n, 0, :]
// ---------------------------------------------------------------------------
__global__ void k_take_cls(const float* __restrict__ x, float* __restrict__ out) {
  long idx = (long)blockIdx.x * 256 + threadIdx.x;   // grid = N*TD/256
  long n = idx / TD_;
  int  j = (int)(idx - n * TD_);
  out[idx] = x[n * (long)S_ * TD_ + j];
}

// ---------------------------------------------------------------------------
// Host orchestration
// ---------------------------------------------------------------------------
extern "C" void kernel_launch(void* const* d_in, const int* in_sizes, int n_in,
                              void* d_out, int out_size, void* d_ws, size_t ws_size,
                              hipStream_t stream) {
  const float* entity     = (const float*)d_in[0];
  const float* cls        = (const float*)d_in[1];
  const float* pair_pos_w = (const float*)d_in[2];
  const float* pos_ln_g   = (const float*)d_in[3];
  const float* pos_ln_b   = (const float*)d_in[4];
  const float* pos_q_w    = (const float*)d_in[5];
  const float* pos_q_b    = (const float*)d_in[6];
  const float* pos_k_w    = (const float*)d_in[7];
  const float* pos_k_b    = (const float*)d_in[8];
  const float* q_w  = (const float*)d_in[9];
  const float* q_b  = (const float*)d_in[10];
  const float* k_w  = (const float*)d_in[11];
  const float* k_b  = (const float*)d_in[12];
  const float* v_w  = (const float*)d_in[13];
  const float* v_b  = (const float*)d_in[14];
  const float* o_w  = (const float*)d_in[15];
  const float* o_b  = (const float*)d_in[16];
  const float* ln1_g = (const float*)d_in[17];
  const float* ln1_b = (const float*)d_in[18];
  const float* fc1_w = (const float*)d_in[19];
  const float* fc1_b = (const float*)d_in[20];
  const float* fc2_w = (const float*)d_in[21];
  const float* fc2_b = (const float*)d_in[22];
  const float* ln2_g = (const float*)d_in[23];
  const float* ln2_b = (const float*)d_in[24];

  char* wp = (char*)d_ws;
  auto alloc = [&](size_t bytes) -> void* {
    void* p = wp;
    wp += (bytes + 255) & ~(size_t)255;
    return p;
  };

  float* x    = (float*)alloc((size_t)M_ * TD_ * 4);
  float* qb_  = (float*)alloc((size_t)M_ * TD_ * 4);
  float* kb_  = (float*)alloc((size_t)M_ * TD_ * 4);
  float* vb_  = (float*)alloc((size_t)M_ * TD_ * 4);
  float* ob_  = (float*)alloc((size_t)M_ * TD_ * 4);
  bf16*  xb   = (bf16*)alloc((size_t)M_ * TD_ * 2);
  bf16*  hb   = (bf16*)alloc((size_t)M_ * FFN_ * 2);
  bf16*  wq_bf  = (bf16*)alloc((size_t)L_ * TD_ * TD_ * 2);
  bf16*  wk_bf  = (bf16*)alloc((size_t)L_ * TD_ * TD_ * 2);
  bf16*  wv_bf  = (bf16*)alloc((size_t)L_ * TD_ * TD_ * 2);
  bf16*  wo_bf  = (bf16*)alloc((size_t)L_ * TD_ * TD_ * 2);
  bf16*  wf1_bf = (bf16*)alloc((size_t)L_ * FFN_ * TD_ * 2);
  bf16*  wf2_bf = (bf16*)alloc((size_t)L_ * TD_ * FFN_ * 2);
  float* wln  = (float*)alloc((size_t)P_ * D_ * 4);
  float* pq   = (float*)alloc((size_t)P_ * TD_ * 4);
  float* pk   = (float*)alloc((size_t)P_ * TD_ * 4);
  float* bias = (float*)alloc((size_t)H_ * S_ * S_ * 4);

  auto cvt = [&](const float* src, bf16* dst, long n) {
    long blocks = (n + 255) / 256;
    if (blocks > 16384) blocks = 16384;
    k_f32_to_bf16<<<(int)blocks, 256, 0, stream>>>(src, dst, n);
  };

  // Weights -> bf16 (all layers at once)
  cvt(q_w,   wq_bf,  (long)L_ * TD_ * TD_);
  cvt(k_w,   wk_bf,  (long)L_ * TD_ * TD_);
  cvt(v_w,   wv_bf,  (long)L_ * TD_ * TD_);
  cvt(o_w,   wo_bf,  (long)L_ * TD_ * TD_);
  cvt(fc1_w, wf1_bf, (long)L_ * FFN_ * TD_);
  cvt(fc2_w, wf2_bf, (long)L_ * TD_ * FFN_);

  // Build x0 (N, S, TD)
  k_build_x<<<(int)((long)N_ * S_ * TD_ / 256), 256, 0, stream>>>(entity, cls, x);

  // Positional bias
  k_wln<<<P_, 256, 0, stream>>>(pair_pos_w, pos_ln_g, pos_ln_b, wln);
  k_posproj<<<2 * P_ * TD_ / 256, 256, 0, stream>>>(wln, pos_q_w, pos_q_b, pos_k_w, pos_k_b,
                                                    pq, pk);
  k_bias<<<H_, 128, 0, stream>>>(pq, pk, bias);

  const dim3 gridTD(TD_ / BN, (int)(M_ / BM));   // (16, 176)
  const dim3 gridFF(FFN_ / BN, (int)(M_ / BM));  // (64, 176)

  for (int l = 0; l < L_; ++l) {
    const bf16* wq = wq_bf + (long)l * TD_ * TD_;
    const bf16* wk = wk_bf + (long)l * TD_ * TD_;
    const bf16* wv = wv_bf + (long)l * TD_ * TD_;
    const bf16* wo = wo_bf + (long)l * TD_ * TD_;
    const bf16* w1 = wf1_bf + (long)l * FFN_ * TD_;
    const bf16* w2 = wf2_bf + (long)l * TD_ * FFN_;

    // q/k/v projections
    cvt(x, xb, M_ * TD_);
    k_gemm_bf16nt<0, 0><<<gridTD, 256, 0, stream>>>(xb, wq, q_b + l * TD_, qb_, TD_, TD_);
    k_gemm_bf16nt<0, 0><<<gridTD, 256, 0, stream>>>(xb, wk, k_b + l * TD_, kb_, TD_, TD_);
    k_gemm_bf16nt<0, 0><<<gridTD, 256, 0, stream>>>(xb, wv, v_b + l * TD_, vb_, TD_, TD_);

    // attention (RoPE + dual softmax + weighted sum)
    k_attention<<<dim3(N_, H_), 128, 0, stream>>>(qb_, kb_, vb_, bias, ob_);

    // output projection + residual + LN1
    cvt(ob_, xb, M_ * TD_);
    k_gemm_bf16nt<0, 0><<<gridTD, 256, 0, stream>>>(xb, wo, o_b + l * TD_, qb_, TD_, TD_);
    k_add_ln<<<(int)M_, 256, 0, stream>>>(x, qb_, ln1_g + l * TD_, ln1_b + l * TD_, x);

    // FFN: h = gelu(x @ fc1^T + b)  [bf16 out], y = h @ fc2^T + b
    cvt(x, xb, M_ * TD_);
    k_gemm_bf16nt<1, 1><<<gridFF, 256, 0, stream>>>(xb, w1, fc1_b + l * FFN_, hb, FFN_, TD_);
    k_gemm_bf16nt<0, 0><<<gridTD, 256, 0, stream>>>(hb, w2, fc2_b + l * TD_, kb_, TD_, FFN_);
    k_add_ln<<<(int)M_, 256, 0, stream>>>(x, kb_, ln2_g + l * TD_, ln2_b + l * TD_, x);
  }

  // out = x[:, 0, :]
  k_take_cls<<<(int)((long)N_ * TD_ / 256), 256, 0, stream>>>(x, (float*)d_out);
}
